// CausalAttention_28750511079559
// MI455X (gfx1250) — matrix operand order repaired and checked by reading, not compiled, not run on hardware
//
#include <hip/hip_runtime.h>
#include <hip/hip_bf16.h>

typedef __bf16 bf16;
typedef __attribute__((ext_vector_type(16))) __bf16 bf16x16;
typedef __attribute__((ext_vector_type(8)))  __bf16 bf16x8;
typedef __attribute__((ext_vector_type(8)))  float  f32x8;
typedef __attribute__((ext_vector_type(4)))  unsigned int u32x4;
typedef __attribute__((ext_vector_type(8)))  int i32x8;
typedef __attribute__((ext_vector_type(4)))  int i32x4;

#define WMMA_BF16(a, b, c) \
  __builtin_amdgcn_wmma_f32_16x16x32_bf16(false, (a), false, (b), (short)0, (c), false, false)

// ---------------------------------------------------------------------------
// Fragment loaders for V_WMMA_F32_16X16X32_BF16 (wave32).
// A (16xK=32, row-major src): lane<16 -> row=lane,   K k0..k0+15 (contiguous)
//                             lane>=16 -> row=lane-16, K k0+16..k0+31
// B (K=32 x 16, src "transposed": row n holds K contiguously):
//                             lane<16 -> col n=lane, K k0..k0+15 (contiguous)
//                             lane>=16 -> col n=lane-16, K k0+16..k0+31
// ---------------------------------------------------------------------------
static __device__ __forceinline__ bf16x16 load_a_frag(const bf16* base, int ld, int row, int k0) {
  const int lane = threadIdx.x & 31;
  const int r    = lane & 15;
  const int koff = (lane >> 4) ? 16 : 0;
  const bf16* p = base + (size_t)(row + r) * ld + (k0 + koff);
  union { bf16x16 v; bf16x8 h[2]; } u;
  u.h[0] = *(const bf16x8*)(p);
  u.h[1] = *(const bf16x8*)(p + 8);
  return u.v;
}

static __device__ __forceinline__ bf16x16 load_b_frag(const bf16* bt, int ld, int n0, int k0) {
  const int lane = threadIdx.x & 31;
  const int n    = n0 + (lane & 15);
  const int koff = (lane >> 4) ? 16 : 0;
  const bf16* p = bt + (size_t)n * ld + (k0 + koff);
  union { bf16x16 v; bf16x8 h[2]; } u;
  u.h[0] = *(const bf16x8*)(p);
  u.h[1] = *(const bf16x8*)(p + 8);
  return u.v;
}

// B fragment out of an LDS tile laid out [64 rows n][32 k] bf16 -> ds_load_b128
static __device__ __forceinline__ bf16x16 load_b_frag_lds(const bf16* tile, int n0) {
  const int lane = threadIdx.x & 31;
  const int n    = n0 + (lane & 15);
  const int koff = (lane >> 4) ? 16 : 0;
  const bf16* p = tile + n * 32 + koff;
  union { bf16x16 v; bf16x8 h[2]; } u;
  u.h[0] = *(const bf16x8*)(p);
  u.h[1] = *(const bf16x8*)(p + 8);
  return u.v;
}

// ---------------------------------------------------------------------------
// Tensor Data Mover: async 2D tile (rows x k, 2-byte elems) global -> LDS.
// D# packing per CDNA5 ISA 8.3/8.4: group0 = {count, lds_addr, global_addr,
// type=2}; group1 = {data_size=2B, tensor dims, tile dims, dim0 stride}.
// ---------------------------------------------------------------------------
static __device__ __forceinline__ void tdm_load_2d_to_lds(
    const void* gaddr, unsigned ldsByteAddr,
    unsigned tile_k, unsigned tile_n,
    unsigned tensor_d0, unsigned tensor_d1, unsigned long long stride0) {
  unsigned long long ga = (unsigned long long)(uintptr_t)gaddr;
  u32x4 g0;
  g0[0] = 1u;                                               // count=1, user D#
  g0[1] = ldsByteAddr;                                      // lds_addr (bytes)
  g0[2] = (unsigned)(ga & 0xFFFFFFFFu);                     // global_addr[31:0]
  g0[3] = (unsigned)((ga >> 32) & 0x1FFFFFFu) | (2u << 30); // [56:32] | type=2
  i32x8 g1;
  g1[0] = (int)(1u << 16);                                  // wg_mask=0, data_size=1 (2B)
  g1[1] = (int)((tensor_d0 & 0xFFFFu) << 16);               // tensor_dim0[15:0]
  g1[2] = (int)((tensor_d0 >> 16) | ((tensor_d1 & 0xFFFFu) << 16));
  g1[3] = (int)((tensor_d1 >> 16) | (tile_k << 16));        // tile_dim0
  g1[4] = (int)(tile_n & 0xFFFFu);                          // tile_dim1, tile_dim2=0
  g1[5] = (int)(unsigned)(stride0 & 0xFFFFFFFFu);           // dim0 stride lo
  g1[6] = (int)(unsigned)((stride0 >> 32) & 0xFFFFu);       // stride hi, dim1_stride=0
  g1[7] = 0;
  i32x4 z4;
  z4[0] = 0; z4[1] = 0; z4[2] = 0; z4[3] = 0;
#if defined(__clang_major__) && __clang_major__ >= 23
  i32x8 z8;
#pragma unroll
  for (int i = 0; i < 8; ++i) z8[i] = 0;
  __builtin_amdgcn_tensor_load_to_lds(g0, g1, z4, z4, z8, 0);
#else
  __builtin_amdgcn_tensor_load_to_lds(g0, g1, z4, z4, 0);
#endif
}

static __device__ __forceinline__ unsigned lds_byte_addr(const void* p) {
  // generic pointers to LDS carry the LDS byte offset in the low 32 bits
  return (unsigned)(uintptr_t)p;
}

// ---------------------------------------------------------------------------
// Precision conversion kernels
// ---------------------------------------------------------------------------
__global__ void cvt_f32_to_bf16(const float* __restrict__ src, bf16* __restrict__ dst, size_t n) {
  size_t i      = (size_t)blockIdx.x * blockDim.x + threadIdx.x;
  size_t stride = (size_t)gridDim.x * blockDim.x;
  for (; i < n; i += stride) dst[i] = (bf16)src[i];
}

// Wt[n][k] = W[k][n], C = 1024 fixed
__global__ void cvt_transpose_1024(const float* __restrict__ W, bf16* __restrict__ Wt) {
  int idx = blockIdx.x * blockDim.x + threadIdx.x;  // exactly 1024*1024 threads
  int n = idx >> 10, k = idx & 1023;
  Wt[idx] = (bf16)W[(size_t)k * 1024 + n];
}

// ---------------------------------------------------------------------------
// GEMM: C[M,N] = A[M,K](bf16) * Bt[N,K]^T(bf16), f32 accumulate.
// Block: 128 thr = 4 waves; block tile 128(M) x 64(N); wave tile 32 x 64.
// B tile (64 n x 32 k) staged in LDS by the TDM, double buffered: DMA of
// tile k+32 overlaps the 8 WMMAs of tile k; A fragment register-pipelined.
// All 4 B fragments loaded as one ds clause before the WMMA chain so the
// compiler can use partial s_wait_dscnt instead of 0-waits per fragment.
// mode 0: bf16 head-split [b,h,t,d] | mode 2: bf16 [b,h,d,t] | mode 3: f32 [M,N]
// ---------------------------------------------------------------------------
__global__ __launch_bounds__(128) void gemm_bf16_wmma(
    const bf16* __restrict__ A, const bf16* __restrict__ Bt, void* __restrict__ Cout,
    int M, int N, int K, int mode) {
  __shared__ bf16 btile[2][64 * 32];  // 2 x 4 KB
  const int wave = threadIdx.x >> 5;
  const int lane = threadIdx.x & 31;
  const int m0 = blockIdx.y * 128 + wave * 32;
  const int n0 = blockIdx.x * 64;

  f32x8 acc[2][4];
#pragma unroll
  for (int mi = 0; mi < 2; ++mi)
#pragma unroll
    for (int j = 0; j < 4; ++j)
#pragma unroll
      for (int r = 0; r < 8; ++r) acc[mi][j][r] = 0.0f;

  // prologue: kick off DMA of first B tile, preload first A fragments
  if (wave == 0)
    tdm_load_2d_to_lds(Bt + (size_t)n0 * K, lds_byte_addr(&btile[0][0]),
                       32, 64, (unsigned)K, (unsigned)N, (unsigned long long)K);
  bf16x16 a_cur[2];
  a_cur[0] = load_a_frag(A, K, m0, 0);
  a_cur[1] = load_a_frag(A, K, m0 + 16, 0);

  for (int k0 = 0; k0 < K; k0 += 32) {
    const int buf = (k0 >> 5) & 1;
    // TENSORcnt is per-wave: non-issuing waves have cnt==0, so this is a
    // cheap unconditional wait (no EXEC save/restore around a branch).
    __builtin_amdgcn_s_wait_tensorcnt(0);
    __syncthreads();
    // overlap: DMA next B tile while this tile is consumed
    if (wave == 0 && k0 + 32 < K)
      tdm_load_2d_to_lds(Bt + (size_t)n0 * K + (k0 + 32),
                         lds_byte_addr(&btile[buf ^ 1][0]),
                         32, 64, (unsigned)K, (unsigned)N, (unsigned long long)K);
    // register-pipeline the A fragments for k+32
    bf16x16 a_nxt[2];
    a_nxt[0] = a_cur[0];
    a_nxt[1] = a_cur[1];
    if (k0 + 32 < K) {
      a_nxt[0] = load_a_frag(A, K, m0, k0 + 32);
      a_nxt[1] = load_a_frag(A, K, m0 + 16, k0 + 32);
      __builtin_prefetch(A + (size_t)(m0 + (lane & 15)) * K + k0 + 96, 0, 1);
    }
    // batch the 4 B-fragment LDS reads, then run the 8-WMMA chain
    bf16x16 bfrag[4];
#pragma unroll
    for (int j = 0; j < 4; ++j) bfrag[j] = load_b_frag_lds(&btile[buf][0], 16 * j);
#pragma unroll
    for (int j = 0; j < 4; ++j) {
      acc[0][j] = WMMA_BF16(a_cur[0], bfrag[j], acc[0][j]);
      acc[1][j] = WMMA_BF16(a_cur[1], bfrag[j], acc[1][j]);
    }
    __syncthreads();  // all reads of btile[buf] done before it is re-filled
    a_cur[0] = a_nxt[0];
    a_cur[1] = a_nxt[1];
  }

  const bool hiHalf = lane >= 16;
  const int nlane = lane & 15;
#pragma unroll
  for (int mi = 0; mi < 2; ++mi) {
    const int mb = m0 + 16 * mi + (hiHalf ? 8 : 0);
    if (mode == 3) {
      float* out = (float*)Cout;
#pragma unroll
      for (int j = 0; j < 4; ++j) {
        int n = n0 + 16 * j + nlane;
#pragma unroll
        for (int r = 0; r < 8; ++r) out[(size_t)(mb + r) * N + n] = acc[mi][j][r];
      }
    } else {
      bf16* out = (bf16*)Cout;
#pragma unroll
      for (int j = 0; j < 4; ++j) {
        int n = n0 + 16 * j + nlane;
        int h = n >> 6, d = n & 63;
#pragma unroll
        for (int r = 0; r < 8; ++r) {
          int m = mb + r;
          int bb = m >> 11, t = m & 2047;
          size_t idx = (mode == 2)
                           ? (((size_t)(bb * 16 + h) * 64 + d) * 2048 + t)
                           : (((size_t)(bb * 16 + h) * 2048 + t) * 64 + d);
          out[idx] = (bf16)acc[mi][j][r];
        }
      }
    }
  }
}

// ---------------------------------------------------------------------------
// Transposed flash attention, ALiBi + causal: S^T = K * Q^T puts q on the
// lane dimension -> softmax and P^T repack need only shfl_xor(16), no LDS.
// K fragments issued as one load clause; V fragments hoisted to the block
// top so their global latency hides behind the softmax VALU/exp work.
// Grid: (T/64, B*nh). Block: 128 thr = 4 waves, wave owns 16 query rows.
// ---------------------------------------------------------------------------
__global__ __launch_bounds__(128) void attn_fwd(
    const bf16* __restrict__ Q, const bf16* __restrict__ Km,
    const bf16* __restrict__ Vt, bf16* __restrict__ O) {
  const int wave = threadIdx.x >> 5;
  const int lane = threadIdx.x & 31;
  const bool hiHalf = lane >= 16;
  const int bh = blockIdx.y;
  const int b = bh >> 4, h = bh & 15;
  const int q0 = blockIdx.x * 64 + wave * 16;

  const bf16* Qbh = Q  + (size_t)bh * 2048 * 64;
  const bf16* Kbh = Km + (size_t)bh * 2048 * 64;
  const bf16* Vbh = Vt + (size_t)bh * 64 * 2048;

  const bf16x16 bq0 = load_b_frag(Qbh, 64, q0, 0);
  const bf16x16 bq1 = load_b_frag(Qbh, 64, q0, 32);

  const float slope = exp2f(-0.5f * (float)(h + 1));  // get_slopes(16)
  const int q = q0 + (lane & 15);
  float m_run = -1e30f, l_run = 0.0f;

  f32x8 acc[4];  // O^T tiles: d = 16t + r(+8 hi), lane = q
#pragma unroll
  for (int t = 0; t < 4; ++t)
#pragma unroll
    for (int r = 0; r < 8; ++r) acc[t][r] = 0.0f;

  f32x8 zc;
#pragma unroll
  for (int r = 0; r < 8; ++r) zc[r] = 0.0f;

  const int kend = q0 + 16;
  for (int j0 = 0; j0 < kend; j0 += 32) {
    // one clause of K fragment loads + V fragments hoisted (used after softmax)
    bf16x16 ak[4];
    ak[0] = load_a_frag(Kbh, 64, j0, 0);
    ak[1] = load_a_frag(Kbh, 64, j0, 32);
    ak[2] = load_a_frag(Kbh, 64, j0 + 16, 0);
    ak[3] = load_a_frag(Kbh, 64, j0 + 16, 32);
    bf16x16 av[4];
#pragma unroll
    for (int t = 0; t < 4; ++t) av[t] = load_a_frag(Vbh, 2048, 16 * t, j0);

    f32x8 s0 = WMMA_BF16(ak[0], bq0, zc);
    s0 = WMMA_BF16(ak[1], bq1, s0);
    f32x8 s1 = WMMA_BF16(ak[2], bq0, zc);
    s1 = WMMA_BF16(ak[3], bq1, s1);

    float p0[8], p1[8];
    float smax = -1e30f;
#pragma unroll
    for (int r = 0; r < 8; ++r) {
      int kk0 = j0 + r + (hiHalf ? 8 : 0);
      int kk1 = kk0 + 16;
      float v0 = s0[r] * 0.125f + slope * (float)(kk0 - 2047);
      float v1 = s1[r] * 0.125f + slope * (float)(kk1 - 2047);
      v0 = (kk0 <= q) ? v0 : -1e30f;
      v1 = (kk1 <= q) ? v1 : -1e30f;
      p0[r] = v0; p1[r] = v1;
      smax = fmaxf(smax, fmaxf(v0, v1));
    }
    smax = fmaxf(smax, __shfl_xor(smax, 16, 32));

    const float mnew = fmaxf(m_run, smax);
    const float corr = __expf(m_run - mnew);
    float lsum = 0.0f;
#pragma unroll
    for (int r = 0; r < 8; ++r) {
      p0[r] = __expf(p0[r] - mnew);
      p1[r] = __expf(p1[r] - mnew);
      lsum += p0[r] + p1[r];
    }
    lsum += __shfl_xor(lsum, 16, 32);
    l_run = l_run * corr + lsum;
    m_run = mnew;
#pragma unroll
    for (int t = 0; t < 4; ++t)
#pragma unroll
      for (int r = 0; r < 8; ++r) acc[t][r] *= corr;

    // pack P^T into B-fragment (32 keys x 16 q) via half-wave exchange
    bf16x16 pf;
#pragma unroll
    for (int r = 0; r < 8; ++r) {
      float o0 = __shfl_xor(p0[r], 16, 32);
      float o1 = __shfl_xor(p1[r], 16, 32);
      pf[r]     = (bf16)(hiHalf ? o1 : p0[r]);
      pf[r + 8] = (bf16)(hiHalf ? p1[r] : o0);
    }

#pragma unroll
    for (int t = 0; t < 4; ++t) acc[t] = WMMA_BF16(av[t], pf, acc[t]);
  }

  const float linv = 1.0f / l_run;
  bf16* Ob = O + ((size_t)b * 2048 + q) * 1024 + h * 64;
#pragma unroll
  for (int t = 0; t < 4; ++t) {
    int dbase = 16 * t + (hiHalf ? 8 : 0);
#pragma unroll
    for (int i = 0; i < 4; ++i) {
      union { bf16 e[2]; unsigned u; } pk;
      pk.e[0] = (bf16)(acc[t][2 * i] * linv);
      pk.e[1] = (bf16)(acc[t][2 * i + 1] * linv);
      *(unsigned*)(Ob + dbase + 2 * i) = pk.u;
    }
  }
}

// ---------------------------------------------------------------------------
extern "C" void kernel_launch(void* const* d_in, const int* in_sizes, int n_in,
                              void* d_out, int out_size, void* d_ws, size_t ws_size,
                              hipStream_t stream) {
  (void)in_sizes; (void)n_in; (void)out_size; (void)ws_size;
  const float* x  = (const float*)d_in[0];
  const float* Wq = (const float*)d_in[1];
  const float* Wk = (const float*)d_in[2];
  const float* Wv = (const float*)d_in[3];
  const float* Wo = (const float*)d_in[4];

  char* w = (char*)d_ws;
  bf16* Xb  = (bf16*)(w + ((size_t)0  << 20));  // 16 MB  x bf16 [8192,1024]
  bf16* Wqt = (bf16*)(w + ((size_t)16 << 20));  //  2 MB  Wq^T bf16
  bf16* Wkt = (bf16*)(w + ((size_t)18 << 20));
  bf16* Wvt = (bf16*)(w + ((size_t)20 << 20));
  bf16* Wot = (bf16*)(w + ((size_t)22 << 20));
  bf16* Qb  = (bf16*)(w + ((size_t)24 << 20));  // 16 MB  [b,h,t,d]
  bf16* Kb  = (bf16*)(w + ((size_t)40 << 20));  // 16 MB  [b,h,t,d]
  bf16* Vtb = (bf16*)(w + ((size_t)56 << 20));  // 16 MB  [b,h,d,t]
  bf16* Ob  = (bf16*)(w + ((size_t)72 << 20));  // 16 MB  [b,t,c]

  cvt_f32_to_bf16<<<2048, 256, 0, stream>>>(x, Xb, (size_t)8192 * 1024);
  cvt_transpose_1024<<<4096, 256, 0, stream>>>(Wq, Wqt);
  cvt_transpose_1024<<<4096, 256, 0, stream>>>(Wk, Wkt);
  cvt_transpose_1024<<<4096, 256, 0, stream>>>(Wv, Wvt);
  cvt_transpose_1024<<<4096, 256, 0, stream>>>(Wo, Wot);

  dim3 gg(16, 64);  // N/64, M/128
  gemm_bf16_wmma<<<gg, 128, 0, stream>>>(Xb, Wqt, Qb,  8192, 1024, 1024, 0);
  gemm_bf16_wmma<<<gg, 128, 0, stream>>>(Xb, Wkt, Kb,  8192, 1024, 1024, 0);
  gemm_bf16_wmma<<<gg, 128, 0, stream>>>(Xb, Wvt, Vtb, 8192, 1024, 1024, 2);

  attn_fwd<<<dim3(32, 64), 128, 0, stream>>>(Qb, Kb, Vtb, Ob);

  gemm_bf16_wmma<<<gg, 128, 0, stream>>>(Ob, Wot, d_out, 8192, 1024, 1024, 3);
}
